// Attention_80805514707533
// MI455X (gfx1250) — matrix operand order repaired and checked by reading, not compiled, not run on hardware
//
#include <hip/hip_runtime.h>
#include <hip/hip_bf16.h>

// Problem constants from the reference: B=512, T=512, C=64, H=128.
#define Bsz 512
#define Tsz 512
#define Csz 64
#define Hsz 128

typedef __bf16 bf16_t;
typedef bf16_t v16bf __attribute__((ext_vector_type(16)));
typedef bf16_t v8bf  __attribute__((ext_vector_type(8)));
typedef float  v8f   __attribute__((ext_vector_type(8)));

#if defined(__has_builtin)
#if __has_builtin(__builtin_amdgcn_tanhf)
#define FAST_TANH(x) __builtin_amdgcn_tanhf(x)   // v_tanh_f32 (gfx1250 trans op)
#endif
#endif
#ifndef FAST_TANH
#define FAST_TANH(x) tanhf(x)
#endif

static __device__ __forceinline__ v8f wmma_bf16(v16bf a, v16bf b, v8f c) {
  // D = A(16x32 bf16) * B(32x16 bf16) + C(16x16 f32)
  return __builtin_amdgcn_wmma_f32_16x16x32_bf16(
      /*neg_a=*/false, a, /*neg_b=*/false, b,
      /*c_mod=*/(short)0, c, /*reuse_a=*/false, /*reuse_b=*/false);
}

// A-fragment (16-bit 16x32 layout): lane m = lane&15, k-half = (lane>>4)*8.
// Elements 0..7  <- K = koff..koff+7   (VGPR0..3, packed pairs)
// Elements 8..15 <- K = koff+16..+23   (VGPR4..7)
static __device__ __forceinline__ v16bf load_a_frag(const bf16_t* S, int row, int koff) {
  v8bf lo = *(const v8bf*)(S + row * Csz + koff);
  v8bf hi = *(const v8bf*)(S + row * Csz + koff + 16);
  return __builtin_shufflevector(lo, hi, 0,1,2,3,4,5,6,7,8,9,10,11,12,13,14,15);
}

// B-fragment built once from global fp32 weights: lane n = lane&15,
// k-half = (lane>>4)*8; B[k][n] = W[h][n][k] -> k 0..7 and k 16..23 from base.
static __device__ __forceinline__ v16bf make_b_frag(const float* Wrow) {
  v16bf r;
  #pragma unroll
  for (int i = 0; i < 16; i += 4) {
    const float4 v = *(const float4*)(Wrow + i + (i >= 8 ? 8 : 0));
    r[i + 0] = (bf16_t)v.x;
    r[i + 1] = (bf16_t)v.y;
    r[i + 2] = (bf16_t)v.z;
    r[i + 3] = (bf16_t)v.w;
  }
  return r;
}

__global__ __launch_bounds__(128) void attn_scan_kernel(
    const float* __restrict__ x,      // [B, T, C]
    const float* __restrict__ w_att,  // [H, C, C]
    const float* __restrict__ w_in,   // [H, C, C]
    const float* __restrict__ w_e,    // [H, C, C]
    float* __restrict__ out)          // [B, T, H]
{
  __shared__ __align__(32) bf16_t sX  [Csz * Csz];  // 8 KB  x_t tile (batch-major)
  __shared__ __align__(32) bf16_t sAtt[Csz * Csz];  // 8 KB  recurrent state
  __shared__ __align__(32) bf16_t sAct[Csz * Csz];  // 8 KB  tanh activations
  __shared__ __align__(32) float  sBuf[Csz * Csz];  // 16 KB pre-softmax logits

  const int tid = threadIdx.x;
  const int h   = blockIdx.x;
  const int b0  = blockIdx.y * 64;

  const int ln    = tid & 31;
  const int wv    = tid >> 5;
  const int lhalf = ln >> 4;             // 0|1
  const int l15   = ln & 15;
  const int mrow  = wv * 16 + l15;       // A-fragment row (batch)
  const int a_off = lhalf * 8;           // A k sub-offset
  const int b_off = lhalf * 8;           // B k sub-offset
  const int drow  = wv * 16 + lhalf * 8; // D rows: drow + r

  // ---- hoist all loop-invariant B fragments into registers (24 x v16bf)
  v16bf fWi[2][4], fWa[2][4], fWe[2][4];
  {
    const size_t off = (size_t)h * (Csz * Csz);
    const float* wiP = w_in  + off;
    const float* waP = w_att + off;
    const float* weP = w_e   + off;
    #pragma unroll
    for (int kt = 0; kt < 2; ++kt) {
      #pragma unroll
      for (int j = 0; j < 4; ++j) {
        const int ro = (j * 16 + l15) * Csz + kt * 32 + b_off;
        fWi[kt][j] = make_b_frag(wiP + ro);
        fWa[kt][j] = make_b_frag(waP + ro);
        fWe[kt][j] = make_b_frag(weP + ro);
      }
    }
  }

  // att state init: 1/C
  for (int i = tid; i < Csz * Csz; i += 128)
    sAtt[i] = (bf16_t)(1.0f / Csz);
  __syncthreads();

  const v8f vzero = {0.f, 0.f, 0.f, 0.f, 0.f, 0.f, 0.f, 0.f};

  for (int t = 0; t < Tsz; ++t) {
    // ---- stage x_t tile (64 batch rows x 64 channels), fp32 -> bf16
    for (int i = tid; i < 64 * 16; i += 128) {
      const int row = i >> 4;
      const int c4  = (i & 15) << 2;
      const float4 v = *(const float4*)(x + ((size_t)(b0 + row) * Tsz + t) * Csz + c4);
      sX[row * Csz + c4 + 0] = (bf16_t)v.x;
      sX[row * Csz + c4 + 1] = (bf16_t)v.y;
      sX[row * Csz + c4 + 2] = (bf16_t)v.z;
      sX[row * Csz + c4 + 3] = (bf16_t)v.w;
    }
    // prefetch next timestep's rows into cache while we compute
    if (t + 1 < Tsz) {
      const int prow = tid >> 1;
      const int pco  = (tid & 1) * 32;
      __builtin_prefetch(x + ((size_t)(b0 + prow) * Tsz + (t + 1)) * Csz + pco, 0, 3);
    }
    __syncthreads();

    // ---- a = x_t @ Wi^T + att @ Wa^T  (16 wmma/wave, B operands in regs)
    v8f acc[4];
    #pragma unroll
    for (int j = 0; j < 4; ++j) acc[j] = vzero;
    #pragma unroll
    for (int kt = 0; kt < 2; ++kt) {
      const v16bf aX = load_a_frag(sX,   mrow, kt * 32 + a_off);
      const v16bf aA = load_a_frag(sAtt, mrow, kt * 32 + a_off);
      #pragma unroll
      for (int j = 0; j < 4; ++j) {
        acc[j] = wmma_bf16(aX, fWi[kt][j], acc[j]);
        acc[j] = wmma_bf16(aA, fWa[kt][j], acc[j]);
      }
    }
    // tanh (hw v_tanh_f32), scatter D-layout -> bf16 A-source tile
    #pragma unroll
    for (int j = 0; j < 4; ++j) {
      #pragma unroll
      for (int r = 0; r < 8; ++r)
        sAct[(drow + r) * Csz + (j * 16 + l15)] = (bf16_t)FAST_TANH(acc[j][r]);
    }
    __syncthreads();

    // ---- e = tanh(a) @ We^T  (8 wmma/wave)
    #pragma unroll
    for (int j = 0; j < 4; ++j) acc[j] = vzero;
    #pragma unroll
    for (int kt = 0; kt < 2; ++kt) {
      const v16bf aT = load_a_frag(sAct, mrow, kt * 32 + a_off);
      #pragma unroll
      for (int j = 0; j < 4; ++j)
        acc[j] = wmma_bf16(aT, fWe[kt][j], acc[j]);
    }
    #pragma unroll
    for (int j = 0; j < 4; ++j) {
      #pragma unroll
      for (int r = 0; r < 8; ++r)
        sBuf[(drow + r) * Csz + (j * 16 + l15)] = acc[j][r];
    }
    __syncthreads();

    // ---- softmax over channels + out[b,t,h] = sum_c a * x_t
    // 2 threads per row (32 contiguous elements each), combine via shfl_xor.
    {
      const int row  = tid >> 1;
      const int co   = (tid & 1) * 32;
      float*        rowp = sBuf + row * Csz + co;
      const bf16_t* xrp  = sX   + row * Csz + co;
      bf16_t*       ap   = sAtt + row * Csz + co;

      float mx = rowp[0];
      #pragma unroll
      for (int c = 1; c < 32; ++c) mx = fmaxf(mx, rowp[c]);
      mx = fmaxf(mx, __shfl_xor(mx, 1));

      float s = 0.0f;
      #pragma unroll
      for (int c = 0; c < 32; ++c) {
        const float e = __expf(rowp[c] - mx);
        rowp[c] = e;
        s += e;
      }
      s += __shfl_xor(s, 1);
      const float inv = 1.0f / s;

      float dot = 0.0f;
      #pragma unroll
      for (int c = 0; c < 32; ++c) {
        const float a = rowp[c] * inv;
        ap[c] = (bf16_t)a;                 // recurrent state for t+1
        dot += a * (float)xrp[c];
      }
      dot += __shfl_xor(dot, 1);
      if ((tid & 1) == 0)
        __builtin_nontemporal_store(dot, out + ((size_t)(b0 + row) * Tsz + t) * Hsz + h);
    }
    __syncthreads();   // sAtt/sX/sBuf consumption complete before next iteration
  }
}

extern "C" void kernel_launch(void* const* d_in, const int* in_sizes, int n_in,
                              void* d_out, int out_size, void* d_ws, size_t ws_size,
                              hipStream_t stream) {
  // setup_inputs order: x, weight_att, weight_input, weight_e (all float32)
  const float* x     = (const float*)d_in[0];
  const float* w_att = (const float*)d_in[1];
  const float* w_in  = (const float*)d_in[2];
  const float* w_e   = (const float*)d_in[3];
  float* out = (float*)d_out;

  dim3 grid(Hsz, Bsz / 64);   // one block per (head, 64-row batch tile)
  attn_scan_kernel<<<grid, 128, 0, stream>>>(x, w_att, w_in, w_e, out);
}
